// LoRA_Llama_Server_34531537059856
// MI455X (gfx1250) — compile-verified, matrix operand order checked by reading, not run on hardware
//
#include <hip/hip_runtime.h>
#include <hip/hip_bf16.h>

typedef __attribute__((ext_vector_type(16))) __bf16 v16bf;
typedef __attribute__((ext_vector_type(8)))  int    v8i;
typedef __attribute__((ext_vector_type(8)))  float  v8f;

typedef __attribute__((ext_vector_type(4))) unsigned int tdm_u32x4;
typedef __attribute__((ext_vector_type(8))) int          tdm_i32x8;
typedef __attribute__((ext_vector_type(4))) int          tdm_i32x4;

#define TILE_M 128
#define TILE_N 128
#define TILE_K 32
#define LDS_STRIDE 40   // bf16 elems; 80B row stride = 20 banks -> conflict-free 16-row gathers

#if defined(__has_builtin)
#if __has_builtin(__builtin_amdgcn_tensor_load_to_lds) && __has_builtin(__builtin_amdgcn_s_wait_tensorcnt)
#define HAVE_TDM 1
#endif
#endif
#ifndef HAVE_TDM
#define HAVE_TDM 0
#endif

#if HAVE_TDM
// Issue one TDM 2-D tile load: tile_d1 rows x tile_d0 elems of 4-byte data,
// row stride stride0 (elems), from global gp (tile start) into LDS byte offset lds_off.
// D# bitfields per CDNA5 ISA 8.3/8.4 (count=1, type=2 "image", data_size=4B).
__device__ __forceinline__ void tdm_load_tile_f32(unsigned lds_off, const float* gp,
                                                  unsigned tensor_d0, unsigned tensor_d1,
                                                  unsigned tile_d0, unsigned tile_d1,
                                                  unsigned stride0)
{
    unsigned long long ga = (unsigned long long)(const void*)gp;
    tdm_u32x4 g0 = {};
    g0[0] = 1u;                                            // count=1 (valid user D#)
    g0[1] = lds_off;                                       // lds_addr [63:32]
    g0[2] = (unsigned)(ga & 0xffffffffu);                  // global_addr[31:0]
    g0[3] = (unsigned)((ga >> 32) & 0x01ffffffu)           // global_addr[56:32]
          | (2u << 30);                                    // type=2
    tdm_i32x8 g1 = {};
    g1[0] = (int)(2u << 16);                               // workgroup_mask=0, data_size=2 (4B)
    g1[1] = (int)((tensor_d0 & 0xffffu) << 16);            // tensor_dim0[15:0] @ bits 63:48
    g1[2] = (int)(((tensor_d0 >> 16) & 0xffffu)            // tensor_dim0[31:16]
          | ((tensor_d1 & 0xffffu) << 16));                // tensor_dim1[15:0]
    g1[3] = (int)(((tensor_d1 >> 16) & 0xffffu)            // tensor_dim1[31:16]
          | ((tile_d0 & 0xffffu) << 16));                  // tile_dim0
    g1[4] = (int)(tile_d1 & 0xffffu);                      // tile_dim1 (tile_dim2=0)
    g1[5] = (int)stride0;                                  // tensor_dim0_stride[31:0]
    g1[6] = 0;                                             // stride0[47:32]=0, dim1_stride lo=0
    g1[7] = 0;
    tdm_i32x4 g2 = {};                                     // 2-D: groups 2/3 unused
    tdm_i32x4 g3 = {};
#if __clang_major__ >= 23
    tdm_i32x8 g4 = {};
    __builtin_amdgcn_tensor_load_to_lds(g0, g1, g2, g3, g4, 0);
#else
    __builtin_amdgcn_tensor_load_to_lds(g0, g1, g2, g3, 0);
#endif
}
#endif

// ---------------------------------------------------------------------------
// GEMM: logits[m][n] = sum_k hidden[m][k] * W[n][k]
// fp32 emulated via bf16 split: a=ahi+alo, b=bhi+blo; ahi*bhi + ahi*blo + alo*bhi
// TDM double-buffers raw fp32 tiles in LDS; convert pass splits into bf16 hi/lo.
// ---------------------------------------------------------------------------
__global__ __launch_bounds__(256)
void lmhead_gemm_bf16x3(const float* __restrict__ A,   // [M, K] hidden (row-major)
                        const float* __restrict__ W,   // [N, K] lm_head weight (row-major)
                        float* __restrict__ C,         // [M, N] logits
                        int M, int N, int K)
{
#if HAVE_TDM
    __shared__ float  rawA[2][TILE_M * TILE_K];   // 2 x 16KB
    __shared__ float  rawB[2][TILE_N * TILE_K];   // 2 x 16KB
#endif
    __shared__ __bf16 Ahi[TILE_M][LDS_STRIDE];
    __shared__ __bf16 Alo[TILE_M][LDS_STRIDE];
    __shared__ __bf16 Bhi[TILE_N][LDS_STRIDE];
    __shared__ __bf16 Blo[TILE_N][LDS_STRIDE];

    const int tid  = threadIdx.x;
    const int lane = tid & 31;
    const int wave = tid >> 5;
    const int wm = wave & 3;        // 4 wave-rows x 32
    const int wn = wave >> 2;       // 2 wave-cols x 64

    // grid.x = M tiles (fast) so co-resident blocks share the same W strip:
    // W streams from HBM once, A (67MB) stays L2-resident.
    const int m0 = blockIdx.x * TILE_M;
    const int n0 = blockIdx.y * TILE_N;

    v8f acc[2][4];
    #pragma unroll
    for (int i = 0; i < 2; ++i)
        #pragma unroll
        for (int j = 0; j < 4; ++j) { v8f z = {}; acc[i][j] = z; }

    const int laneRow = lane & 15;
    // 16-bit A/B fragment K-layout (ISA 7.12.2): lanes 0-15 hold K {0..7,16..23},
    // lanes 16-31 hold K {8..15,24..31}  -> two 8-elem (16B) chunks per lane.
    const int laneKa = (lane >> 4) << 3;      // 0 or 8

#if HAVE_TDM
    // Prologue: wave 0 kicks TDM for k-tile 0 (EXEC-independent, TENSORcnt-tracked).
    if (wave == 0) {
        tdm_load_tile_f32((unsigned)(unsigned long long)(void*)&rawA[0][0],
                          A + (size_t)m0 * K, K, M, TILE_K, TILE_M, K);
        tdm_load_tile_f32((unsigned)(unsigned long long)(void*)&rawB[0][0],
                          W + (size_t)n0 * K, K, N, TILE_K, TILE_N, K);
    }
#endif

    int buf = 0;
    for (int k0 = 0; k0 < K; k0 += TILE_K, buf ^= 1) {
#if HAVE_TDM
        if (wave == 0) {
            __builtin_amdgcn_s_wait_tensorcnt(0);     // raw[buf] tiles have landed
            if (k0 + TILE_K < K) {                    // prefetch next k-tile via TDM
                tdm_load_tile_f32((unsigned)(unsigned long long)(void*)&rawA[buf ^ 1][0],
                                  A + (size_t)m0 * K + k0 + TILE_K, K, M, TILE_K, TILE_M, K);
                tdm_load_tile_f32((unsigned)(unsigned long long)(void*)&rawB[buf ^ 1][0],
                                  W + (size_t)n0 * K + k0 + TILE_K, K, N, TILE_K, TILE_N, K);
            }
        }
#endif
        __syncthreads();   // raw[buf] visible; split buffers free (prev frags loaded)

        // ---- convert: fp32 -> split bf16 hi/lo in LDS ----
        // 128 rows x 32 cols = 1024 float4 per tile; 256 threads x 4 each
        #pragma unroll
        for (int i = 0; i < 4; ++i) {
            int idx = tid + i * 256;
            int r   = idx >> 3;
            int c4  = (idx & 7) << 2;
#if HAVE_TDM
            const float4 av = *(const float4*)&rawA[buf][idx << 2];
            const float4 bv = *(const float4*)&rawB[buf][idx << 2];
#else
            const float4 av = *(const float4*)(A + (size_t)(m0 + r) * K + k0 + c4);
            const float4 bv = *(const float4*)(W + (size_t)(n0 + r) * K + k0 + c4);
#endif
            float af[4] = {av.x, av.y, av.z, av.w};
            float bf[4] = {bv.x, bv.y, bv.z, bv.w};
            #pragma unroll
            for (int j = 0; j < 4; ++j) {
                __bf16 ah = (__bf16)af[j];
                __bf16 bh = (__bf16)bf[j];
                Ahi[r][c4 + j] = ah;
                Alo[r][c4 + j] = (__bf16)(af[j] - (float)ah);
                Bhi[r][c4 + j] = bh;
                Blo[r][c4 + j] = (__bf16)(bf[j] - (float)bh);
            }
        }
        __syncthreads();

        // ---- gather WMMA fragments from LDS ----
        v16bf fah[2], fal[2], fbh[4], fbl[4];
        #pragma unroll
        for (int mi = 0; mi < 2; ++mi) {
            int r = wm * 32 + mi * 16 + laneRow;
            {
                int4 c0 = *(const int4*)&Ahi[r][laneKa];
                int4 c1 = *(const int4*)&Ahi[r][laneKa + 16];
                v8i t; t[0]=c0.x; t[1]=c0.y; t[2]=c0.z; t[3]=c0.w;
                       t[4]=c1.x; t[5]=c1.y; t[6]=c1.z; t[7]=c1.w;
                fah[mi] = __builtin_bit_cast(v16bf, t);
            }
            {
                int4 c0 = *(const int4*)&Alo[r][laneKa];
                int4 c1 = *(const int4*)&Alo[r][laneKa + 16];
                v8i t; t[0]=c0.x; t[1]=c0.y; t[2]=c0.z; t[3]=c0.w;
                       t[4]=c1.x; t[5]=c1.y; t[6]=c1.z; t[7]=c1.w;
                fal[mi] = __builtin_bit_cast(v16bf, t);
            }
        }
        #pragma unroll
        for (int ni = 0; ni < 4; ++ni) {
            int c = wn * 64 + ni * 16 + laneRow;
            {
                int4 c0 = *(const int4*)&Bhi[c][laneKa];
                int4 c1 = *(const int4*)&Bhi[c][laneKa + 16];
                v8i t; t[0]=c0.x; t[1]=c0.y; t[2]=c0.z; t[3]=c0.w;
                       t[4]=c1.x; t[5]=c1.y; t[6]=c1.z; t[7]=c1.w;
                fbh[ni] = __builtin_bit_cast(v16bf, t);
            }
            {
                int4 c0 = *(const int4*)&Blo[c][laneKa];
                int4 c1 = *(const int4*)&Blo[c][laneKa + 16];
                v8i t; t[0]=c0.x; t[1]=c0.y; t[2]=c0.z; t[3]=c0.w;
                       t[4]=c1.x; t[5]=c1.y; t[6]=c1.z; t[7]=c1.w;
                fbl[ni] = __builtin_bit_cast(v16bf, t);
            }
        }

        // ---- 24 WMMAs per K-tile: hi*hi + hi*lo + lo*hi, fp32 accumulate ----
        #pragma unroll
        for (int mi = 0; mi < 2; ++mi)
            #pragma unroll
            for (int ni = 0; ni < 4; ++ni) {
                acc[mi][ni] = __builtin_amdgcn_wmma_f32_16x16x32_bf16(
                    false, fah[mi], false, fbh[ni], (short)0, acc[mi][ni], false, false);
                acc[mi][ni] = __builtin_amdgcn_wmma_f32_16x16x32_bf16(
                    false, fah[mi], false, fbl[ni], (short)0, acc[mi][ni], false, false);
                acc[mi][ni] = __builtin_amdgcn_wmma_f32_16x16x32_bf16(
                    false, fal[mi], false, fbh[ni], (short)0, acc[mi][ni], false, false);
            }
    }

    // ---- epilogue: C/D layout — VGPR j: M=j (lanes 0-15) / M=8+j (lanes 16-31), N=lane&15 ----
    const int outN = n0 + wn * 64;
    #pragma unroll
    for (int mi = 0; mi < 2; ++mi) {
        int mBase = m0 + wm * 32 + mi * 16 + ((lane >> 4) << 3);
        #pragma unroll
        for (int ni = 0; ni < 4; ++ni) {
            int n = outN + ni * 16 + laneRow;
            #pragma unroll
            for (int j = 0; j < 8; ++j)
                C[(size_t)(mBase + j) * N + n] = acc[mi][ni][j];
        }
    }
}

// ---------------------------------------------------------------------------
// Per-row cross-entropy: nll[row] = logsumexp(logits[row,:]) - logits[row,label]
// One block per shifted row (deterministic two-pass, no atomics).
// ---------------------------------------------------------------------------
__global__ __launch_bounds__(256)
void ce_row_nll(const float* __restrict__ logits, const int* __restrict__ labels,
                float* __restrict__ nll, int S, int V)
{
    const int row = blockIdx.x;          // 0 .. B*(S-1)-1
    const int b = row / (S - 1);
    const int s = row - b * (S - 1);
    const float* x = logits + (size_t)(b * S + s) * V;
    const int label = labels[b * S + s + 1];

    __shared__ float red[256];
    const int tid = threadIdx.x;

    float mx = -3.402823466e+38f;
    for (int v = tid; v < V; v += 256) mx = fmaxf(mx, x[v]);
    red[tid] = mx; __syncthreads();
    #pragma unroll
    for (int off = 128; off > 0; off >>= 1) {
        if (tid < off) red[tid] = fmaxf(red[tid], red[tid + off]);
        __syncthreads();
    }
    mx = red[0]; __syncthreads();

    float sum = 0.f;
    for (int v = tid; v < V; v += 256) sum += __expf(x[v] - mx);
    red[tid] = sum; __syncthreads();
    #pragma unroll
    for (int off = 128; off > 0; off >>= 1) {
        if (tid < off) red[tid] += red[tid + off];
        __syncthreads();
    }
    if (tid == 0) nll[row] = logf(red[0]) + mx - x[label];
}

__global__ __launch_bounds__(256)
void ce_mean(const float* __restrict__ nll, float* __restrict__ out, int n)
{
    __shared__ float red[256];
    const int tid = threadIdx.x;
    float s = 0.f;
    for (int i = tid; i < n; i += 256) s += nll[i];
    red[tid] = s; __syncthreads();
    #pragma unroll
    for (int off = 128; off > 0; off >>= 1) {
        if (tid < off) red[tid] += red[tid + off];
        __syncthreads();
    }
    if (tid == 0) out[0] = red[0] / (float)n;
}

// ---------------------------------------------------------------------------
extern "C" void kernel_launch(void* const* d_in, const int* in_sizes, int n_in,
                              void* d_out, int out_size, void* d_ws, size_t ws_size,
                              hipStream_t stream)
{
    (void)in_sizes; (void)n_in; (void)out_size; (void)ws_size;

    const float* hidden = (const float*)d_in[0];   // [B, S, D] fp32
    const float* W      = (const float*)d_in[1];   // [V, D]    fp32
    const int*   labels = (const int*)d_in[2];     // [B, S]    int32

    float* out    = (float*)d_out;   // out[0] = loss, out[1..] = logits [B,S,V]
    float* logits = out + 1;
    float* nll    = (float*)d_ws;    // B*(S-1) floats of scratch

    const int B = 2, S = 2048, D = 4096, V = 32000;
    const int M = B * S;

    dim3 grid(M / TILE_M, V / TILE_N);   // 32 x 250; M-tiles fast => W strip shared in L2
    lmhead_gemm_bf16x3<<<grid, 256, 0, stream>>>(hidden, W, logits, M, V, D);

    const int rows = B * (S - 1);
    ce_row_nll<<<rows, 256, 0, stream>>>(logits, labels, nll, S, V);
    ce_mean<<<1, 256, 0, stream>>>(nll, out, rows);
}
